// Generator_1752346657047
// MI455X (gfx1250) — matrix operand order repaired and checked by reading, not compiled, not run on hardware
//
#include <hip/hip_runtime.h>
#include <hip/hip_bf16.h>

typedef __attribute__((ext_vector_type(16))) _Float16 v16h;
typedef __attribute__((ext_vector_type(8)))  _Float16 v8h;
typedef __attribute__((ext_vector_type(2)))  _Float16 v2h;
typedef __attribute__((ext_vector_type(8)))  float    v8f;

#define TPB 256
#define MAXADJ 1024

__device__ __forceinline__ float leakyf(float v){ return v >= 0.f ? v : 0.1f * v; }

// ---------------------------------------------------------------------------
// Weight pre-swizzle: OIHW f32 -> B-operand panels for v_wmma_f32_16x16x32_f16
// Layout: [kt][ct][lane(0..31)][e(0..15)] f16, where
//   K = kt*32 + 8*(lane>>4) + e + (e>=8?8:0),  co = ct*16 + (lane&15)
// so each lane loads its 16 B-halves as one contiguous 32B v16h load.
// ---------------------------------------------------------------------------
__global__ void k_prep_w(const float* __restrict__ w, _Float16* __restrict__ wp,
                         int cout, int cin, int ksz, int nKt, int nCt)
{
    long t = (long)blockIdx.x * TPB + threadIdx.x;
    long total = (long)nKt * nCt * 512;
    if (t >= total) return;
    int e    = (int)(t & 15);
    int lane = (int)((t >> 4) & 31);
    long g   = t >> 9;               // kt*nCt + ct
    int ct   = (int)(g % nCt);
    int kt   = (int)(g / nCt);
    int lg = lane >> 4, lr = lane & 15;
    int K  = kt * 32 + 8 * lg + e + ((e & 8) ? 8 : 0);
    int co = ct * 16 + lr;
    int kk2 = ksz * ksz;
    int Ktot = cin * kk2;
    float v = 0.f;
    if (K < Ktot && co < cout) {
        int ci  = K / kk2;
        int rem = K - ci * kk2;      // r*ksz + s
        v = w[(size_t)(co * cin + ci) * kk2 + rem];
    }
    wp[t] = (_Float16)v;
}

// ---------------------------------------------------------------------------
// Implicit-GEMM conv via WMMA with asynchronous LDS staging.
//  - block = (image n = blockIdx.y, 8 consecutive 16-pixel M tiles): stages all
//    cin rows (+halo) into LDS via GLOBAL_LOAD_ASYNC_TO_LDS_B128 (ASYNCcnt),
//    one s_wait_asynccnt 0 before the barrier.
//  - K -> (ci,dy,dx) decomposition precomputed into an LDS int2 table.
//  - each wave computes one 16-pixel M tile x ALL cout tiles (nCt<=3):
//    A gathered from LDS once per K-step, up to 3 back-to-back WMMAs.
// A layout (f16 16x32): lanes0-15 M=lane, K=e+(e>=8?8:0); lanes16-31 +8.
// C/D (f32 16x16): M = vgpr + 8*(lane>>4), N = lane&15.
// op: 0=bias raw, 1=bias+leaky, 2=bias+skip+leaky, 3=bias+tanh -> f32 (co==0)
// ---------------------------------------------------------------------------
__global__ __launch_bounds__(TPB)
void k_conv_wmma(const _Float16* __restrict__ in, const _Float16* __restrict__ wp,
                 const float* __restrict__ bias,
                 _Float16* __restrict__ out, const _Float16* __restrict__ skip,
                 float* __restrict__ out32,
                 int cin, int cout, int H, int W, int OH, int OW,
                 int lW, int lOW,                   // log2(W), log2(OW)
                 int stride, int ksz, int pad, int nKt, int nCt,
                 long inBs, long outBs, long skipBs, int op)
{
    __shared__ __align__(16) _Float16 sh[16384];   // staged input tile (<= 32KB)
    __shared__ int2 tab[512];                      // K -> {ldsOff, (dy<<16)|(dx&0xffff)}

    int n  = blockIdx.y;
    int bt = blockIdx.x;
    int OHW = OH * OW;
    int nMt = OHW >> 4;

    int pix0   = bt * 128;
    int pixEnd = pix0 + 128 < OHW ? pix0 + 128 : OHW;
    int y0 = pix0 >> lOW;
    int y1 = (pixEnd - 1) >> lOW;
    int ylo = y0 * stride - pad;           if (ylo < 0) ylo = 0;
    int yhi = y1 * stride + ksz - 1 - pad; if (yhi > H - 1) yhi = H - 1;
    int rows = yhi - ylo + 1;

    const _Float16* inb = in + (size_t)n * inBs;

    // async-stage rows [ylo,yhi] x all cin: 16B per lane-op, direct to LDS
    int rw = rows << lW;                  // rows * W, multiple of 8
    int totalOct = (cin * rw) >> 3;
    for (int i8 = threadIdx.x; i8 < totalOct; i8 += TPB) {
        int idx = i8 << 3;
        int ci = idx / rw;
        int r  = idx - ci * rw;
        int ry = r >> lW;
        int ix = r & (W - 1);
        const _Float16* g = inb + ((size_t)ci * H + (ylo + ry)) * W + ix;
        unsigned lds = (unsigned)(size_t)&sh[idx];  // low 32b = LDS offset
        asm volatile("global_load_async_to_lds_b128 %0, %1, off"
                     :: "v"(lds), "v"(g) : "memory");
    }
    // K decomposition table
    int kk2  = ksz * ksz;
    int Ktot = cin * kk2;
    for (int K = threadIdx.x; K < nKt * 32; K += TPB) {
        if (K < Ktot) {
            int ci  = K / kk2;
            int rem = K - ci * kk2;
            int rr  = rem / ksz;
            int dx  = rem - rr * ksz - pad;
            int dy  = rr - pad;
            tab[K] = make_int2(((ci * rows + dy) << lW) + dx, (dy << 16) | (dx & 0xffff));
        } else {
            tab[K] = make_int2(0, 0x40000000);     // dy huge -> always out of range
        }
    }
    asm volatile("s_wait_asynccnt 0x0" ::: "memory");
    __syncthreads();

    int w8 = threadIdx.x >> 5;
    int mt = bt * 8 + w8;
    if (mt >= nMt) return;                // wave-uniform, after the only barrier

    int lane = threadIdx.x & 31, lg = lane >> 4, lr = lane & 15;
    int Mrow = mt * 16 + lr;
    int y = Mrow >> lOW, x = Mrow & (OW - 1);
    int ys = y * stride, xs = x * stride;
    int pixBase = ((ys - ylo) << lW) + xs;

    v8f acc[3];
    v8f z8 = {0.f,0.f,0.f,0.f,0.f,0.f,0.f,0.f};
    acc[0] = z8; acc[1] = z8; acc[2] = z8;

    for (int kt = 0; kt < nKt; ++kt) {
        int Kb = kt * 32 + 8 * lg;
        v16h a;
        #pragma unroll
        for (int e = 0; e < 16; ++e) {
            int K   = Kb + e + ((e & 8) ? 8 : 0);
            int2 t2 = tab[K];
            int dy = t2.y >> 16;
            int dx = (int)(short)(t2.y & 0xffff);
            int iy = ys + dy;
            int ix = xs + dx;
            _Float16 v = (_Float16)0.f;
            if ((unsigned)iy < (unsigned)H && (unsigned)ix < (unsigned)W)
                v = sh[pixBase + t2.x];
            a[e] = v;
        }
        #pragma unroll
        for (int c2 = 0; c2 < 3; ++c2) {
            if (c2 < nCt) {
                v16h b = *(const v16h*)(wp + (((size_t)kt * nCt + c2) * 32 + lane) * 16);
                acc[c2] = __builtin_amdgcn_wmma_f32_16x16x32_f16(false, a, false, b,
                                                                 (short)0, acc[c2],
                                                                 false, false);
            }
        }
    }

    // vectorized epilogue: per lane, the 8 outputs of a cout tile are contiguous
    #pragma unroll
    for (int c2 = 0; c2 < 3; ++c2) {
        if (c2 < nCt) {
            int co = c2 * 16 + lr;
            bool valid = co < cout;
            float bb = valid ? bias[co] : 0.f;
            if (op == 3) {
                if (lr == 0 && c2 == 0) {
                    v8f o;
                    #pragma unroll
                    for (int r2 = 0; r2 < 8; ++r2) o[r2] = tanhf(acc[0][r2] + bb);
                    *(v8f*)(out32 + (size_t)n * OHW + mt * 16 + 8 * lg) = o;
                }
            } else if (valid) {
                size_t boff = (size_t)n * outBs + (size_t)co * OHW + mt * 16 + 8 * lg;
                v8h sk;
                if (op == 2)
                    sk = *(const v8h*)(skip + (size_t)n * skipBs + (size_t)co * OHW + mt * 16 + 8 * lg);
                v8h o;
                #pragma unroll
                for (int r2 = 0; r2 < 8; ++r2) {
                    float v = acc[c2][r2] + bb;
                    if (op == 2) v += (float)sk[r2];
                    if (op >= 1) v = leakyf(v);
                    o[r2] = (_Float16)v;
                }
                *(v8h*)(out + boff) = o;
            }
        }
    }
}

// ---------------------------------------------------------------------------
// Linear: x = [z|given_y] @ W^T + b ; M=256 (batch), N=512, K=146 (padded 160)
// Output written directly into channels 0..7 of the 16-channel fcat tensor.
// ---------------------------------------------------------------------------
__global__ __launch_bounds__(TPB)
void k_linear_wmma(const float* __restrict__ z, const float* __restrict__ gy,
                   const _Float16* __restrict__ wp, const float* __restrict__ bias,
                   _Float16* __restrict__ fcat, int nKt)
{
    int wave = blockIdx.x * 8 + (threadIdx.x >> 5);   // exactly 512 waves launched
    int lane = threadIdx.x & 31, lg = lane >> 4, lr = lane & 15;
    const int nCt = 32;
    int mt = wave / nCt, ct = wave % nCt;
    int n = mt * 16 + lr;

    v8f acc = {0.f,0.f,0.f,0.f,0.f,0.f,0.f,0.f};
    for (int kt = 0; kt < nKt; ++kt) {
        v16h b = *(const v16h*)(wp + (((size_t)kt * nCt + ct) * 32 + lane) * 16);
        v16h a;
        int Kb = kt * 32 + 8 * lg;
        #pragma unroll
        for (int e = 0; e < 16; ++e) {
            int K = Kb + e + ((e & 8) ? 8 : 0);
            float v = 0.f;
            if (K < 128)      v = z[(size_t)n * 128 + K];
            else if (K < 146) v = gy[(size_t)n * 18 + (K - 128)];
            a[e] = (_Float16)v;
        }
        acc = __builtin_amdgcn_wmma_f32_16x16x32_f16(false, a, false, b,
                                                     (short)0, acc, false, false);
    }
    int j = ct * 16 + lr;
    float bb = bias[j];
    int c = j >> 6, pix = j & 63;
    #pragma unroll
    for (int r2 = 0; r2 < 8; ++r2) {
        int n2 = mt * 16 + r2 + 8 * lg;
        fcat[((size_t)n2 * 16 + c) * 64 + pix] = (_Float16)(acc[r2] + bb);
    }
}

// ---------------------------------------------------------------------------
// Instance norm (per n,c over HW), optional relu, in place. One block per (n,c).
// ---------------------------------------------------------------------------
__global__ __launch_bounds__(TPB)
void k_inorm(_Float16* __restrict__ x, int HW, int relu)
{
    __shared__ float ss[TPB], ss2[TPB];
    size_t base = (size_t)blockIdx.x * HW;
    float s = 0.f, s2 = 0.f;
    for (int i = threadIdx.x; i < HW; i += TPB) {
        float v = (float)x[base + i]; s += v; s2 += v * v;
    }
    ss[threadIdx.x] = s; ss2[threadIdx.x] = s2; __syncthreads();
    for (int w = TPB / 2; w > 0; w >>= 1) {
        if ((int)threadIdx.x < w) { ss[threadIdx.x] += ss[threadIdx.x + w];
                                    ss2[threadIdx.x] += ss2[threadIdx.x + w]; }
        __syncthreads();
    }
    float m = ss[0] / HW;
    float var = ss2[0] / HW - m * m;
    float r = rsqrtf(var + 1e-5f);
    for (int i = threadIdx.x; i < HW; i += TPB) {
        float v = ((float)x[base + i] - m) * r;
        if (relu) v = fmaxf(v, 0.f);
        x[base + i] = (_Float16)v;
    }
}

// nearest-neighbour 2x upsample: (N,C,S,S) -> (N,C,2S,2S); one v2h store per
// output pair (both halves of a pair share the same source pixel).
__global__ void k_up2(const _Float16* __restrict__ in, _Float16* __restrict__ out,
                      int S, long totalPairs)
{
    long i2 = (long)blockIdx.x * TPB + threadIdx.x;
    if (i2 >= totalPairs) return;
    long i = i2 << 1;
    int S2 = 2 * S;
    int x = (int)(i % S2);           // even
    long t = i / S2;
    int y = (int)(t % S2);
    long nc = t / S2;
    _Float16 h = in[(size_t)nc * S * S + (size_t)(y >> 1) * S + (x >> 1)];
    v2h o; o[0] = h; o[1] = h;
    *(v2h*)&out[i] = o;
}

// out = leaky(down2(a) + down2(b)); a,b: (N,C,2S,2S); out channel-offset capable
__global__ void k_down2_add_leaky(const _Float16* __restrict__ a,
                                  const _Float16* __restrict__ b,
                                  _Float16* __restrict__ outBase,
                                  int C, int S, long outBs, long total)
{
    long i = (long)blockIdx.x * TPB + threadIdx.x;
    if (i >= total) return;
    int SS = S * S;
    int pix = (int)(i % SS);
    int c   = (int)((i / SS) % C);
    int n   = (int)(i / ((long)SS * C));
    int y = pix / S, x = pix - y * S;
    int S2 = 2 * S;
    size_t ib = (((size_t)(n * C + c) * S2) + 2 * y) * S2 + 2 * x;
    float sum = (float)a[ib] + (float)a[ib + 1] + (float)a[ib + S2] + (float)a[ib + S2 + 1]
              + (float)b[ib] + (float)b[ib + 1] + (float)b[ib + S2] + (float)b[ib + S2 + 1];
    float v = leakyf(sum * 0.25f);
    outBase[(size_t)n * outBs + (size_t)c * SS + pix] = (_Float16)v;
}

// ---------------------------------------------------------------------------
// Deterministic signed graph pooling (gather form).
// k_build_adj: one thread per node scans edges in fixed order -> adjacency list
//   entry = otherNode | (sign>0 ? 0x100 : 0)
// k_pool_pack: per (n, element): sums feats[other] over incident edges into the
// two signed pools and writes enc_in = concat([feats, pool+, pool-]) directly.
// ---------------------------------------------------------------------------
__global__ void k_build_adj(const int* __restrict__ ew, int E,
                            int* __restrict__ adj, int* __restrict__ cnts)
{
    int v = blockIdx.x * blockDim.x + threadIdx.x;
    if (v >= 256) return;
    int c = 0;
    for (int e = 0; e < E; ++e) {
        int src = ew[e * 3 + 0];
        int sg  = ew[e * 3 + 1];
        int dst = ew[e * 3 + 2];
        int tag = (sg > 0) ? 0x100 : 0;
        if (dst == v && c < MAXADJ) adj[v * MAXADJ + c++] = src | tag;  // feats[src] -> dst
        if (src == v && c < MAXADJ) adj[v * MAXADJ + c++] = dst | tag;  // feats[dst] -> src
    }
    cnts[v] = c;
}

__global__ __launch_bounds__(TPB)
void k_pool_pack(const _Float16* __restrict__ X,
                 const int* __restrict__ adj, const int* __restrict__ cnts,
                 _Float16* __restrict__ enc, long CHW, long total)
{
    long i = (long)blockIdx.x * TPB + threadIdx.x;
    if (i >= total) return;
    int n  = (int)(i / CHW);
    long r = i % CHW;
    float sp = 0.f, sn = 0.f;
    int c = cnts[n];
    const int* al = adj + (size_t)n * MAXADJ;
    for (int j = 0; j < c; ++j) {            // wave-uniform loop (n uniform)
        int ent = al[j];
        float v = (float)X[(size_t)(ent & 0xff) * CHW + r];
        if (ent & 0x100) sp += v; else sn += v;
    }
    size_t eb = (size_t)n * 3 * CHW;
    enc[eb + r]            = X[i];
    enc[eb + CHW + r]      = (_Float16)sp;
    enc[eb + 2 * CHW + r]  = (_Float16)sn;
}

__global__ void k_f32_to_f16(const float* __restrict__ in, _Float16* __restrict__ out, long total)
{
    long i = (long)blockIdx.x * TPB + threadIdx.x;
    if (i < total) out[i] = (_Float16)in[i];
}

// ---------------------------------------------------------------------------
// Host orchestration
// ---------------------------------------------------------------------------
struct WP { _Float16* p; int nKt, nCt; };

static inline int ilog2(int v){ int l = 0; while ((1 << l) < v) ++l; return l; }

extern "C" void kernel_launch(void* const* d_in, const int* in_sizes, int n_in,
                              void* d_out, int out_size, void* d_ws, size_t ws_size,
                              hipStream_t stream)
{
    (void)in_sizes; (void)n_in; (void)out_size; (void)ws_size;
    const int N = 256;
    auto F  = [&](int i) -> const float* { return (const float*)d_in[i]; };
    const int* ew = (const int*)d_in[3];

    // ---- workspace bump allocator ----
    char* wsb = (char*)d_ws; size_t off = 0;
    auto alloc = [&](size_t bytes) -> char* {
        char* p = wsb + off;
        off = (off + bytes + 255) & ~((size_t)255);
        return p;
    };
    const size_t EL48 = (size_t)N * 48 * 64 * 64;   // 48ch @ 64^2
    const size_t EL16 = (size_t)N * 16 * 64 * 64;   // 16ch @ 64^2
    _Float16* AR_A = (_Float16*)alloc(EL48 * 2);
    _Float16* AR_B = (_Float16*)alloc(EL48 * 2);
    _Float16* AR_C = (_Float16*)alloc(EL48 * 2);
    _Float16* XB   = (_Float16*)alloc(EL16 * 2);
    _Float16* YB   = (_Float16*)alloc(EL16 * 2);
    _Float16* pin  = (_Float16*)alloc((size_t)N * 2 * 4096 * 2);
    _Float16* fcat = (_Float16*)alloc((size_t)N * 16 * 64 * 2);
    _Float16* E0   = (_Float16*)alloc((size_t)N * 8 * 1024 * 2);
    _Float16* E1   = (_Float16*)alloc((size_t)N * 8 * 1024 * 2);
    _Float16* E2   = (_Float16*)alloc((size_t)N * 8 * 1024 * 2);
    _Float16* E3   = (_Float16*)alloc((size_t)N * 8 * 1024 * 2);
    _Float16* E4   = (_Float16*)alloc((size_t)N * 8 * 256 * 2);
    int* adj  = (int*)alloc((size_t)N * MAXADJ * 4);
    int* cnts = (int*)alloc((size_t)N * 4);

    auto nb = [](long t) -> int { return (int)((t + TPB - 1) / TPB); };

    // ---- weight pre-swizzle ----
    auto prep = [&](const void* w, int cout, int cin, int ksz) -> WP {
        int kk = ksz * ksz, Ktot = cin * kk;
        int nKt = (Ktot + 31) / 32, nCt = (cout + 15) / 16;
        size_t el = (size_t)nKt * nCt * 512;
        _Float16* wp = (_Float16*)alloc(el * 2);
        k_prep_w<<<nb((long)el), TPB, 0, stream>>>((const float*)w, wp, cout, cin, ksz, nKt, nCt);
        return {wp, nKt, nCt};
    };

    WP lw   = prep(d_in[4], 512, 146, 1);
    WP e0w  = prep(d_in[6], 8, 2, 3);
    WP r1w1 = prep(d_in[8], 8, 8, 3),  r1w2 = prep(d_in[10], 8, 8, 3),  r1ws = prep(d_in[12], 8, 8, 1);
    WP r2w1 = prep(d_in[14], 8, 8, 3), r2w2 = prep(d_in[16], 8, 8, 3),  r2ws = prep(d_in[18], 8, 8, 1);
    WP uw[3][3];
    for (int u = 0; u < 3; ++u) {
        int b = 20 + 6 * u;
        uw[u][0] = prep(d_in[b + 0], 16, 16, 3);
        uw[u][1] = prep(d_in[b + 2], 16, 16, 3);
        uw[u][2] = prep(d_in[b + 4], 16, 16, 1);
    }
    WP hw[4] = { prep(d_in[38], 16, 16, 3), prep(d_in[40], 16, 16, 3),
                 prep(d_in[42], 16, 16, 3), prep(d_in[44], 16, 16, 3) };
    WP hcw = prep(d_in[46], 1, 16, 3);
    WP cw[4][5];
    for (int i = 0; i < 4; ++i) {
        int b = 48 + 10 * i;
        cw[i][0] = prep(d_in[b + 0], 48, 48, 3);
        cw[i][1] = prep(d_in[b + 2], 48, 48, 3);
        cw[i][2] = prep(d_in[b + 4], 48, 48, 3);
        cw[i][3] = prep(d_in[b + 6], 48, 48, 3);
        cw[i][4] = prep(d_in[b + 8], 16, 48, 3);
    }

    // ---- conv launcher ----
    auto conv = [&](const _Float16* in, WP w, const float* bias,
                    int cin, int cout, int H, int Wd, int stride, int ksz, int pad,
                    _Float16* out, long inBs, long outBs,
                    int op, const _Float16* skip, long skipBs, float* out32) {
        int OH = (H + 2 * pad - ksz) / stride + 1;
        int OW = (Wd + 2 * pad - ksz) / stride + 1;
        int nMt = (OH * OW) >> 4;
        int tilesPerN = (nMt + 7) / 8;
        dim3 grid(tilesPerN, N);
        k_conv_wmma<<<grid, TPB, 0, stream>>>(in, w.p, bias, out, skip, out32,
            cin, cout, H, Wd, OH, OW, ilog2(Wd), ilog2(OW),
            stride, ksz, pad, w.nKt, w.nCt, inBs, outBs, skipBs, op);
    };

    // ================= forward =================
    k_build_adj<<<1, TPB, 0, stream>>>(ew, 512, adj, cnts);
    // given_p -> f16
    k_f32_to_f16<<<nb((long)N * 2 * 4096), TPB, 0, stream>>>(F(2), pin, (long)N * 2 * 4096);
    // l1 linear -> fcat channels 0..7
    k_linear_wmma<<<64, TPB, 0, stream>>>(F(0), F(1), lw.p, F(5), fcat, lw.nKt);

    // encoder: conv stride2 + inorm + relu
    conv(pin, e0w, F(7), 2, 8, 64, 64, 2, 3, 1, E0, 2L * 4096, 8L * 1024, 0, nullptr, 0, nullptr);
    k_inorm<<<N * 8, TPB, 0, stream>>>(E0, 1024, 1);
    // enc_rb1 (down) @32 -> 16
    conv(E0, r1w1, F(9),  8, 8, 32, 32, 1, 3, 1, E1, 8L * 1024, 8L * 1024, 1, nullptr, 0, nullptr);
    conv(E1, r1w2, F(11), 8, 8, 32, 32, 1, 3, 1, E2, 8L * 1024, 8L * 1024, 0, nullptr, 0, nullptr);
    conv(E0, r1ws, F(13), 8, 8, 32, 32, 1, 1, 0, E3, 8L * 1024, 8L * 1024, 0, nullptr, 0, nullptr);
    k_down2_add_leaky<<<nb((long)N * 8 * 256), TPB, 0, stream>>>(E2, E3, E4, 8, 16, 8L * 256, (long)N * 8 * 256);
    // enc_rb2 (down) @16 -> 8, write into fcat channels 8..15
    conv(E4, r2w1, F(15), 8, 8, 16, 16, 1, 3, 1, E1, 8L * 256, 8L * 256, 1, nullptr, 0, nullptr);
    conv(E1, r2w2, F(17), 8, 8, 16, 16, 1, 3, 1, E2, 8L * 256, 8L * 256, 0, nullptr, 0, nullptr);
    conv(E4, r2ws, F(19), 8, 8, 16, 16, 1, 1, 0, E3, 8L * 256, 8L * 256, 0, nullptr, 0, nullptr);
    k_down2_add_leaky<<<nb((long)N * 8 * 64), TPB, 0, stream>>>(E2, E3, fcat + 8 * 64, 8, 8, 16L * 64, (long)N * 8 * 64);

    // ---- cmp stage ----
    auto cmpStage = [&](int ci, const _Float16* X, int S, _Float16* Y) {
        long CHW = 16L * S * S, tot = (long)N * CHW;
        k_pool_pack<<<nb(tot), TPB, 0, stream>>>(X, adj, cnts, AR_A, CHW, tot);
        long BS = 48L * S * S;
        int b = 48 + 10 * ci;
        conv(AR_A, cw[ci][0], F(b + 1), 48, 48, S, S, 1, 3, 1, AR_B, BS, BS, 1, nullptr, 0, nullptr);
        conv(AR_B, cw[ci][1], F(b + 3), 48, 48, S, S, 1, 3, 1, AR_C, BS, BS, 2, AR_A, BS, nullptr);
        conv(AR_C, cw[ci][2], F(b + 5), 48, 48, S, S, 1, 3, 1, AR_B, BS, BS, 1, nullptr, 0, nullptr);
        conv(AR_B, cw[ci][3], F(b + 7), 48, 48, S, S, 1, 3, 1, AR_A, BS, BS, 2, AR_C, BS, nullptr);
        conv(AR_A, cw[ci][4], F(b + 9), 48, 16, S, S, 1, 3, 1, Y, BS, 16L * S * S, 0, nullptr, 0, nullptr);
        k_inorm<<<N * 16, TPB, 0, stream>>>(Y, S * S, 1);
    };

    // ---- up stage (resblock 'up') ----
    auto upStage = [&](int ui, const _Float16* X, int S, _Float16* out) {
        int S2 = 2 * S;
        long C16 = 16L * S2 * S2;
        k_up2<<<nb((long)N * C16 / 2), TPB, 0, stream>>>(X, AR_A, S, (long)N * C16 / 2);
        int b = 20 + 6 * ui;
        conv(AR_A, uw[ui][2], F(b + 5), 16, 16, S2, S2, 1, 1, 0, AR_B, C16, C16, 0, nullptr, 0, nullptr);
        conv(AR_A, uw[ui][0], F(b + 1), 16, 16, S2, S2, 1, 3, 1, AR_C, C16, C16, 1, nullptr, 0, nullptr);
        conv(AR_C, uw[ui][1], F(b + 3), 16, 16, S2, S2, 1, 3, 1, out, C16, C16, 2, AR_B, C16, nullptr);
    };

    cmpStage(0, fcat, 8,  YB);
    upStage(0, YB, 8,  XB);
    cmpStage(1, XB, 16, YB);
    upStage(1, YB, 16, XB);
    cmpStage(2, XB, 32, YB);
    upStage(2, YB, 32, XB);
    cmpStage(3, XB, 64, YB);

    // ---- head ----
    long C16 = 16L * 4096;
    conv(YB,   hw[0], F(39), 16, 16, 64, 64, 1, 3, 1, AR_B, C16, C16, 1, nullptr, 0, nullptr);
    conv(AR_B, hw[1], F(41), 16, 16, 64, 64, 1, 3, 1, AR_A, C16, C16, 2, YB, C16, nullptr);
    conv(AR_A, hw[2], F(43), 16, 16, 64, 64, 1, 3, 1, AR_B, C16, C16, 1, nullptr, 0, nullptr);
    conv(AR_B, hw[3], F(45), 16, 16, 64, 64, 1, 3, 1, AR_C, C16, C16, 2, AR_A, C16, nullptr);
    conv(AR_C, hcw,  F(47), 16, 1, 64, 64, 1, 3, 1, nullptr, C16, 0, 3, nullptr, 0, (float*)d_out);
}
